// GAT_71408126263810
// MI455X (gfx1250) — matrix operand order repaired and checked by reading, not compiled, not run on hardware
//
#include <hip/hip_runtime.h>
#include <hip/hip_bf16.h>

// ---------------------------------------------------------------------------
// GAT (3-layer) forward for MI455X (gfx1250).
//   - hs = x @ Ws via v_wmma_f32_16x16x32_bf16 (bf16 in, f32 accumulate)
//   - a_d computed via algebraic fold (Wd@ad) -> tiny GEMV, hd never built
//   - segment softmax with native u32 atomic-max (order-preserving float map)
//     and f32 atomic adds; aggregation with coalesced f32 atomics.
// ---------------------------------------------------------------------------

typedef __attribute__((ext_vector_type(16))) __bf16 v16bf;
typedef __attribute__((ext_vector_type(8)))  __bf16 v8bf;
typedef __attribute__((ext_vector_type(8)))  float  v8f;

#define NN   50000
#define KKF  256      // every layer's in-dim and out-dim is 256 (H*C = H*OUT = IN)
#define HH   4
#define NEGS 0.2f

// order-preserving float <-> uint map (unsigned compare == float compare)
__device__ __forceinline__ unsigned f2ord(float f) {
  unsigned b = __float_as_uint(f);
  return (b & 0x80000000u) ? ~b : (b | 0x80000000u);
}
__device__ __forceinline__ float ord2f(unsigned u) {
  return __uint_as_float((u & 0x80000000u) ? (u & 0x7FFFFFFFu) : ~u);
}

// ---- fp32 -> bf16 convert -------------------------------------------------
__global__ void gat_cvt_bf16(const float* __restrict__ in, __bf16* __restrict__ out, int n) {
  int i = blockIdx.x * 256 + threadIdx.x;
  if (i < n) out[i] = (__bf16)in[i];
}

// ---- weight transpose + convert: wt[nout,kin] = W[kin,nout]  (256x256) ----
__global__ void gat_wt_transpose(const float* __restrict__ w, __bf16* __restrict__ wt) {
  int i = blockIdx.x * 256 + threadIdx.x;   // 65536 elements
  int nout = i >> 8, kin = i & 255;
  wt[i] = (__bf16)w[kin * 256 + nout];
}

// ---- fold attention vectors: vs[k,h] = sum_c Ws[k,h*64+c]*as[h,c] ---------
__global__ void gat_fold(const float* __restrict__ Ws, const float* __restrict__ Wd,
                         const float* __restrict__ as_, const float* __restrict__ ad_,
                         float* __restrict__ vs, float* __restrict__ vd) {
  int i = blockIdx.x * 256 + threadIdx.x;   // 1024 elements
  if (i >= 256 * HH) return;
  int k = i >> 2, h = i & 3;
  float s = 0.f, d = 0.f;
  for (int c = 0; c < 64; ++c) {
    s += Ws[k * 256 + h * 64 + c] * as_[h * 64 + c];
    d += Wd[k * 256 + h * 64 + c] * ad_[h * 64 + c];
  }
  vs[k * 4 + h] = s;
  vd[k * 4 + h] = d;
}

// ---- WMMA GEMM: y[N,256] = xb[N,256](bf16) @ wt^T  (wt is [256,256] = W^T) -
// 8 waves per block; each wave computes a 16x64 strip with 4 accumulators.
__global__ void __launch_bounds__(256)
gat_gemm_wmma(const __bf16* __restrict__ xb, const __bf16* __restrict__ wt,
              float* __restrict__ y) {
  const int wave  = threadIdx.x >> 5;
  const int lane  = threadIdx.x & 31;
  const int m0    = (blockIdx.x * 8 + wave) * 16;
  if (m0 >= NN) return;                      // wave-uniform: EXEC stays all-1s
  const int n0    = blockIdx.y * 64;
  const int mrow  = lane & 15;               // A: row; B/D: column
  const int khalf = lane >> 4;               // K-half select

  v8f acc0 = {}, acc1 = {}, acc2 = {}, acc3 = {};

  #pragma unroll
  for (int kb = 0; kb < 8; ++kb) {           // K = 256 = 8 * 32
    // A fragment (16x32 bf16): lane holds row m0+mrow,
    //   elems 0..7  -> K = kb*32 + khalf*16 + (0..7)
    //   elems 8..15 -> K = kb*32 + khalf*16 + 8 + (0..7)
    const v8bf* pa = (const v8bf*)(xb + (size_t)(m0 + mrow) * 256 + kb * 32 + khalf * 16);
    v8bf lo = pa[0];
    v8bf hi = pa[1];                         // +8 bf16
    v16bf a;
    #pragma unroll
    for (int i = 0; i < 8; ++i) { a[i] = lo[i]; a[i + 8] = hi[i]; }

    // B fragments (32x16 bf16): lane holds column n, K = kb*32 + khalf*16 + (0..15)
    const __bf16* wb = wt + kb * 32 + khalf * 16;
    v16bf b0 = *(const v16bf*)(wb + (size_t)(n0 +  0 + mrow) * 256);
    v16bf b1 = *(const v16bf*)(wb + (size_t)(n0 + 16 + mrow) * 256);
    v16bf b2 = *(const v16bf*)(wb + (size_t)(n0 + 32 + mrow) * 256);
    v16bf b3 = *(const v16bf*)(wb + (size_t)(n0 + 48 + mrow) * 256);

    acc0 = __builtin_amdgcn_wmma_f32_16x16x32_bf16(false, a, false, b0, (short)0, acc0, false, false);
    acc1 = __builtin_amdgcn_wmma_f32_16x16x32_bf16(false, a, false, b1, (short)0, acc1, false, false);
    acc2 = __builtin_amdgcn_wmma_f32_16x16x32_bf16(false, a, false, b2, (short)0, acc2, false, false);
    acc3 = __builtin_amdgcn_wmma_f32_16x16x32_bf16(false, a, false, b3, (short)0, acc3, false, false);
  }

  // D layout: VGPR r holds row m0 + r + 8*khalf, column n0 + mrow (+16t)
  #pragma unroll
  for (int r = 0; r < 8; ++r) {
    size_t row = (size_t)(m0 + r + 8 * khalf) * 256;
    y[row + n0 +  0 + mrow] = acc0[r];
    y[row + n0 + 16 + mrow] = acc1[r];
    y[row + n0 + 32 + mrow] = acc2[r];
    y[row + n0 + 48 + mrow] = acc3[r];
  }
}

// ---- attention scores: a_s[n,h] = x[n,:]@vs[:,h]; a_d likewise ------------
__global__ void gat_ascore(const __bf16* __restrict__ xb, const float* __restrict__ vs,
                           const float* __restrict__ vd,
                           float* __restrict__ a_s, float* __restrict__ a_d) {
  int n = blockIdx.x * 256 + threadIdx.x;
  if (n >= NN) return;
  float s[4] = {0, 0, 0, 0}, d[4] = {0, 0, 0, 0};
  for (int k = 0; k < 256; ++k) {
    float xv = (float)xb[(size_t)n * 256 + k];
    #pragma unroll
    for (int h = 0; h < 4; ++h) {
      s[h] += xv * vs[k * 4 + h];
      d[h] += xv * vd[k * 4 + h];
    }
  }
  #pragma unroll
  for (int h = 0; h < 4; ++h) { a_s[n * 4 + h] = s[h]; a_d[n * 4 + h] = d[h]; }
}

// ---- zero per-layer accumulators ------------------------------------------
__global__ void gat_init(float* __restrict__ hnext, unsigned* __restrict__ amax,
                         float* __restrict__ denom) {
  int i = blockIdx.x * 256 + threadIdx.x;
  if (i < NN * 256) hnext[i] = 0.f;
  if (i < NN * 4) { amax[i] = 0u; denom[i] = 0.f; }
}

// ---- edge pass 1: alpha + segment max -------------------------------------
__global__ void gat_edge_alpha(const int* __restrict__ src, const int* __restrict__ dst,
                               const float* __restrict__ a_s, const float* __restrict__ a_d,
                               float* __restrict__ alpha, unsigned* __restrict__ amax, int ne4) {
  int i = blockIdx.x * 256 + threadIdx.x;
  if (i >= ne4) return;
  int e = i >> 2, h = i & 3;
  int s = src[e], d = dst[e];
  float al = a_s[s * 4 + h] + a_d[d * 4 + h];
  al = al > 0.f ? al : NEGS * al;
  alpha[i] = al;
  atomicMax(&amax[d * 4 + h], f2ord(al));
}

// ---- edge pass 2: exp + segment sum ---------------------------------------
__global__ void gat_edge_exp(const int* __restrict__ dst, float* __restrict__ alpha,
                             const unsigned* __restrict__ amax, float* __restrict__ denom,
                             int ne4) {
  int i = blockIdx.x * 256 + threadIdx.x;
  if (i >= ne4) return;
  int e = i >> 2, h = i & 3;
  int d = dst[e];
  float ex = __expf(alpha[i] - ord2f(amax[d * 4 + h]));
  alpha[i] = ex;                 // overwrite in place with exp value
  atomicAdd(&denom[d * 4 + h], ex);
}

// ---- edge pass 3: weighted aggregation (block per edge, 256 channels) -----
__global__ void __launch_bounds__(256)
gat_edge_agg(const int* __restrict__ src, const int* __restrict__ dst,
             const float* __restrict__ hs, const float* __restrict__ ex,
             const float* __restrict__ denom, float* __restrict__ out) {
  int e = blockIdx.x;
  int j = threadIdx.x;           // 0..255 = h*64 + c
  int s = src[e], d = dst[e];    // uniform -> scalar loads
  int h = j >> 6;
  float w = ex[(size_t)e * 4 + h] / (denom[d * 4 + h] + 1e-16f);
  atomicAdd(&out[(size_t)d * 256 + j], hs[(size_t)s * 256 + j] * w);
}

// ---- bias + ELU, emit bf16 input for next layer ---------------------------
__global__ void gat_elu_bias(const float* __restrict__ hnext, const float* __restrict__ b,
                             __bf16* __restrict__ xb) {
  int i = blockIdx.x * 256 + threadIdx.x;
  if (i >= NN * 256) return;
  float v = hnext[i] + b[i & 255];
  v = v > 0.f ? v : (__expf(v) - 1.f);
  xb[i] = (__bf16)v;
}

// ---- layer 2 tail: head-mean + bias + log_softmax (64 classes) ------------
__global__ void __launch_bounds__(64)
gat_finalize(const float* __restrict__ hnext, const float* __restrict__ b2,
             float* __restrict__ out) {
  int n = blockIdx.x;
  int j = threadIdx.x;           // 0..63
  size_t base = (size_t)n * 256;
  float v = 0.25f * (hnext[base + j] + hnext[base + 64 + j] +
                     hnext[base + 128 + j] + hnext[base + 192 + j]) + b2[j];
  __shared__ float sm[64];
  sm[j] = v;
  __syncthreads();
  float m = -3.4e38f;
  for (int i = 0; i < 64; ++i) m = fmaxf(m, sm[i]);
  __syncthreads();
  sm[j] = __expf(v - m);
  __syncthreads();
  float s = 0.f;
  for (int i = 0; i < 64; ++i) s += sm[i];
  out[(size_t)n * 64 + j] = v - m - __logf(s);
}

// ---------------------------------------------------------------------------
extern "C" void kernel_launch(void* const* d_in, const int* in_sizes, int n_in,
                              void* d_out, int out_size, void* d_ws, size_t ws_size,
                              hipStream_t stream) {
  const float* x   = (const float*)d_in[0];
  const int*   ei  = (const int*)d_in[1];
  const int    ne  = in_sizes[1] / 2;          // edges incl. self-loops
  const int*   src = ei;
  const int*   dst = ei + ne;

  const float* Ws[3] = { (const float*)d_in[2],  (const float*)d_in[7],  (const float*)d_in[12] };
  const float* Wd[3] = { (const float*)d_in[3],  (const float*)d_in[8],  (const float*)d_in[13] };
  const float* As[3] = { (const float*)d_in[4],  (const float*)d_in[9],  (const float*)d_in[14] };
  const float* Ad[3] = { (const float*)d_in[5],  (const float*)d_in[10], (const float*)d_in[15] };
  const float* Bs[3] = { (const float*)d_in[6],  (const float*)d_in[11], (const float*)d_in[16] };

  // workspace carve-out (256B aligned)
  char* ws = (char*)d_ws;
  size_t o = 0;
  auto carve = [&](size_t bytes) -> void* {
    void* p = ws + o;
    o += (bytes + 255) & ~(size_t)255;
    return p;
  };
  __bf16*   wt    = (__bf16*)  carve((size_t)256 * 256 * 2);     // transposed bf16 weights
  __bf16*   xb    = (__bf16*)  carve((size_t)NN * 256 * 2);      // layer input, bf16
  float*    hs    = (float*)   carve((size_t)NN * 256 * 4);      // x @ Ws
  float*    hnext = (float*)   carve((size_t)NN * 256 * 4);      // aggregated output
  float*    a_s   = (float*)   carve((size_t)NN * 4 * 4);
  float*    a_d   = (float*)   carve((size_t)NN * 4 * 4);
  unsigned* amax  = (unsigned*)carve((size_t)NN * 4 * 4);
  float*    denom = (float*)   carve((size_t)NN * 4 * 4);
  float*    alpha = (float*)   carve((size_t)ne * 4 * 4);        // alpha -> exp in place
  float*    vs    = (float*)   carve((size_t)256 * 4 * 4);
  float*    vd    = (float*)   carve((size_t)256 * 4 * 4);

  const int ne4       = ne * 4;
  const int blkNe4    = (ne4 + 255) / 256;
  const int blkNF     = (NN * 256 + 255) / 256;                  // 50000
  const int blkNodes  = (NN + 255) / 256;
  const dim3 gemmGrid((NN / 16 + 7) / 8, 4, 1);                  // 391 x 4

  // x -> bf16 once
  gat_cvt_bf16<<<blkNF, 256, 0, stream>>>(x, xb, NN * 256);

  for (int L = 0; L < 3; ++L) {
    gat_wt_transpose<<<256, 256, 0, stream>>>(Ws[L], wt);
    gat_fold<<<4, 256, 0, stream>>>(Ws[L], Wd[L], As[L], Ad[L], vs, vd);
    gat_gemm_wmma<<<gemmGrid, 256, 0, stream>>>(xb, wt, hs);
    gat_ascore<<<blkNodes, 256, 0, stream>>>(xb, vs, vd, a_s, a_d);
    gat_init<<<blkNF, 256, 0, stream>>>(hnext, amax, denom);
    gat_edge_alpha<<<blkNe4, 256, 0, stream>>>(src, dst, a_s, a_d, alpha, amax, ne4);
    gat_edge_exp<<<blkNe4, 256, 0, stream>>>(dst, alpha, amax, denom, ne4);
    gat_edge_agg<<<ne, 256, 0, stream>>>(src, dst, hs, alpha, denom, hnext);
    if (L < 2) {
      gat_elu_bias<<<blkNF, 256, 0, stream>>>(hnext, Bs[L], xb);
    } else {
      gat_finalize<<<NN, 64, 0, stream>>>(hnext, Bs[L], (float*)d_out);
    }
  }
}
